// OlmoAttention_90117003804824
// MI455X (gfx1250) — compile-verified, matrix operand order checked by reading, not run on hardware
//
#include <hip/hip_runtime.h>
#include <hip/hip_bf16.h>

// ---------------------------------------------------------------------------
// OLMo attention layer on gfx1250 (MI455X), bf16 WMMA + fp32 accumulation.
// All WMMA operands are PRE-SWIZZLED in operand layout so every lane's
// 16 bf16 elements are 32 contiguous bytes -> ds_load_b128 / global_load_b128.
//
// Operand element mapping (wave32, 16-bit data, 16x32 A / 32x16 B tiles):
//   value (i, k)  ->  lane = (i&15) + 16*(((k&31)>>3)&1)
//                     elem = ((k&31)&7) + 8*((k&31)>>4)
// One tile = 32 lanes x 16 elems = 512 bf16 = 1 KB, stored contiguously.
//
// Staging uses CDNA5 async global->LDS copies (ASYNCcnt) when the builtin is
// available; otherwise a register-pipelined copy (loads issued before the
// WMMA block, stores after) so global latency always overlaps compute.
// ---------------------------------------------------------------------------

typedef __attribute__((ext_vector_type(16))) __bf16 bf16x16;
typedef __attribute__((ext_vector_type(8)))  float  f32x8;
typedef int v4i_ __attribute__((vector_size(16)));   // matches builtin param

constexpr int   B_   = 2;
constexpr int   S_   = 2048;
constexpr int   HID_ = 2048;
constexpr int   NH_  = 16;
constexpr int   HD_  = 128;
constexpr float SCALE_ = 0.08838834764831845f;   // 128^-0.5
constexpr float CLIP_  = 8.0f;
constexpr float NEG_   = -1.0e30f;

#if defined(__has_builtin)
#if __has_builtin(__builtin_amdgcn_global_load_async_to_lds_b128)
#define HAVE_ASYNC_LDS 1
#endif
#endif
#ifndef HAVE_ASYNC_LDS
#define HAVE_ASYNC_LDS 0
#endif

#if HAVE_ASYNC_LDS
__device__ __forceinline__ void async_cp16(const __bf16* g, __bf16* l) {
  __builtin_amdgcn_global_load_async_to_lds_b128(
      (__attribute__((address_space(1))) v4i_*)g,
      (__attribute__((address_space(3))) v4i_*)l, 0, 0);
}
#endif

__device__ __forceinline__ void wait_async0() {
#if defined(__has_builtin)
#if __has_builtin(__builtin_amdgcn_s_wait_asynccnt)
  __builtin_amdgcn_s_wait_asynccnt(0);
#else
  asm volatile("s_wait_asynccnt 0x0" ::: "memory");
#endif
#else
  asm volatile("s_wait_asynccnt 0x0" ::: "memory");
#endif
}

__device__ __forceinline__ f32x8 wmma_bf16(bf16x16 a, bf16x16 b, f32x8 c) {
  return __builtin_amdgcn_wmma_f32_16x16x32_bf16(
      false, a, false, b, (short)0, c, false, false);
}

__device__ __forceinline__ int op_lane(int i, int k) {
  return (i & 15) + ((((k & 31) >> 3) & 1) << 4);
}
__device__ __forceinline__ int op_elem(int k) {
  return ((k & 31) & 7) + (((k & 31) >> 4) << 3);
}

// ---------------------------------------------------------------------------
// Cast fp32 row-major [R][C] -> bf16 A-operand tiles [rt = r/16][kb = c/32].
// ---------------------------------------------------------------------------
__global__ void swizzle_opA_kernel(const float* __restrict__ in,
                                   __bf16* __restrict__ out, int R, int C) {
  const int KB = C >> 5;
  const int total = R * C;
  const int stride = gridDim.x * blockDim.x;
  for (int idx = blockIdx.x * blockDim.x + threadIdx.x; idx < total; idx += stride) {
    const int r = idx / C, c = idx - r * C;
    const size_t dst = ((size_t)((r >> 4) * KB + (c >> 5)) << 9)
                     + op_lane(r, c) * 16 + op_elem(c);
    out[dst] = (__bf16)in[idx];
  }
}

// ---------------------------------------------------------------------------
// Cast fp32 row-major weight [K][N] -> bf16 B-operand tiles [nt = n/16][kb = k/32].
// ---------------------------------------------------------------------------
__global__ void swizzle_opB_kernel(const float* __restrict__ in,
                                   __bf16* __restrict__ out, int K, int N) {
  const int KB = K >> 5;
  const int total = K * N;
  const int stride = gridDim.x * blockDim.x;
  for (int idx = blockIdx.x * blockDim.x + threadIdx.x; idx < total; idx += stride) {
    const int k = idx / N, n = idx - k * N;
    const size_t dst = ((size_t)((n >> 4) * KB + (k >> 5)) << 9)
                     + op_lane(n, k) * 16 + op_elem(k);
    out[dst] = (__bf16)in[idx];
  }
}

// ---------------------------------------------------------------------------
// bf16 WMMA GEMM: C(f32 MxN) = A x B, operands pre-swizzled.
// Block = 128 threads (4 waves), tile 128x128; wave = 64x64 (16 WMMA/K-step).
// Double-buffered LDS with async (or register-pipelined) staging.
// ---------------------------------------------------------------------------
__global__ __launch_bounds__(128, 1)
void gemm_bf16_wmma(const __bf16* __restrict__ Aop,
                    const __bf16* __restrict__ Bop,
                    float* __restrict__ C,
                    int M, int N, int K, int doClip) {
  __shared__ __align__(32) __bf16 ldsA[2][8 * 512];   // 8 row-tiles / buffer
  __shared__ __align__(32) __bf16 ldsB[2][8 * 512];   // 8 col-tiles / buffer

  const int t    = threadIdx.x;
  const int wv   = t >> 5;
  const int lane = t & 31;
  const int n0   = blockIdx.x * 128;
  const int m0   = blockIdx.y * 128;
  const int KB   = K >> 5;
  const int mtB  = m0 >> 4;
  const int ntB  = n0 >> 4;

  f32x8 acc[16];
#pragma unroll
  for (int i = 0; i < 16; ++i) acc[i] = (f32x8){};

  // ---- initial stage of kb=0 into buffer 0 ----
#if HAVE_ASYNC_LDS
#pragma unroll
  for (int i = 0; i < 4; ++i) {
    const int c = t + 128 * i;
    const int tile = c >> 6, off = c & 63;
    async_cp16((const __bf16*)(((const float4*)Aop) + (size_t)((mtB + tile) * KB) * 64 + off),
               (__bf16*)(((float4*)ldsA[0]) + c));
    async_cp16((const __bf16*)(((const float4*)Bop) + (size_t)((ntB + tile) * KB) * 64 + off),
               (__bf16*)(((float4*)ldsB[0]) + c));
  }
  wait_async0();
#else
#pragma unroll
  for (int i = 0; i < 4; ++i) {
    const int c = t + 128 * i;
    const int tile = c >> 6, off = c & 63;
    ((float4*)ldsA[0])[c] = ((const float4*)Aop)[(size_t)((mtB + tile) * KB) * 64 + off];
    ((float4*)ldsB[0])[c] = ((const float4*)Bop)[(size_t)((ntB + tile) * KB) * 64 + off];
  }
#endif
  __syncthreads();

  for (int kb = 0; kb < KB; ++kb) {
    const int cur = kb & 1, nxt = cur ^ 1;
    const bool more = (kb + 1 < KB);

#if HAVE_ASYNC_LDS
    if (more) {            // fire-and-forget async copies for the next panel
#pragma unroll
      for (int i = 0; i < 4; ++i) {
        const int c = t + 128 * i;
        const int tile = c >> 6, off = c & 63;
        async_cp16((const __bf16*)(((const float4*)Aop) + (size_t)((mtB + tile) * KB + kb + 1) * 64 + off),
                   (__bf16*)(((float4*)ldsA[nxt]) + c));
        async_cp16((const __bf16*)(((const float4*)Bop) + (size_t)((ntB + tile) * KB + kb + 1) * 64 + off),
                   (__bf16*)(((float4*)ldsB[nxt]) + c));
      }
    }
#else
    float4 ra[4], rb[4];   // issue loads now; they fly during the WMMAs
    if (more) {
#pragma unroll
      for (int i = 0; i < 4; ++i) {
        const int c = t + 128 * i;
        const int tile = c >> 6, off = c & 63;
        ra[i] = ((const float4*)Aop)[(size_t)((mtB + tile) * KB + kb + 1) * 64 + off];
        rb[i] = ((const float4*)Bop)[(size_t)((ntB + tile) * KB + kb + 1) * 64 + off];
      }
    }
#endif
    if (kb + 2 < KB) {     // speculative prefetch two steps ahead
      __builtin_prefetch(Aop + (size_t)((mtB + (t >> 4)) * KB + kb + 2) * 512 + (t & 15) * 32, 0, 1);
      __builtin_prefetch(Bop + (size_t)((ntB + (t >> 4)) * KB + kb + 2) * 512 + (t & 15) * 32, 0, 1);
    }

    // ---- 16 WMMA on the current buffer ----
    bf16x16 a[4];
#pragma unroll
    for (int it = 0; it < 4; ++it)
      a[it] = *(const bf16x16*)&ldsA[cur][(((wv & 1) * 4 + it) << 9) + lane * 16];
#pragma unroll
    for (int jt = 0; jt < 4; ++jt) {
      const bf16x16 b =
          *(const bf16x16*)&ldsB[cur][(((wv >> 1) * 4 + jt) << 9) + lane * 16];
#pragma unroll
      for (int it = 0; it < 4; ++it)
        acc[it * 4 + jt] = wmma_bf16(a[it], b, acc[it * 4 + jt]);
    }

#if HAVE_ASYNC_LDS
    if (more) wait_async0();          // copies landed before the barrier
#else
    if (more) {
#pragma unroll
      for (int i = 0; i < 4; ++i) {
        const int c = t + 128 * i;
        ((float4*)ldsA[nxt])[c] = ra[i];
        ((float4*)ldsB[nxt])[c] = rb[i];
      }
    }
#endif
    __syncthreads();
  }

#pragma unroll
  for (int it = 0; it < 4; ++it)
#pragma unroll
    for (int jt = 0; jt < 4; ++jt)
#pragma unroll
      for (int r = 0; r < 8; ++r) {
        const int m = m0 + (wv & 1) * 64 + it * 16 + r + ((lane >> 4) << 3);
        const int n = n0 + (wv >> 1) * 64 + jt * 16 + (lane & 15);
        float v = acc[it * 4 + jt][r];
        if (doClip) v = fminf(fmaxf(v, -CLIP_), CLIP_);
        C[(size_t)m * N + n] = v;
      }
}

// ---------------------------------------------------------------------------
// RoPE + split; emits Q/K as per-head operand tiles [st = s/16][db = d/32] and
// V as per-head B-operand tiles [kb = s/32][jt = d/16].
// ---------------------------------------------------------------------------
__global__ void rope_split_kernel(const float* __restrict__ qkv,
                                  const int* __restrict__ pos,
                                  __bf16* __restrict__ Qop,
                                  __bf16* __restrict__ Kop,
                                  __bf16* __restrict__ Vop) {
  const int tid = blockIdx.x * blockDim.x + threadIdx.x;
  if (tid >= B_ * S_ * NH_ * 64) return;
  const int d   = tid & 63;
  const int h   = (tid >> 6) & (NH_ - 1);
  const int tok = tid >> 10;
  const int b = tok / S_, s = tok % S_;

  const float p    = (float)pos[s];
  const float freq = __powf(10000.0f, -(float)d * (1.0f / 64.0f));
  float sn, cs;
  __sincosf(p * freq, &sn, &cs);

  const size_t rb = (size_t)tok * (3 * HID_) + h * HD_ + d;
  const float q1 = qkv[rb],            q2 = qkv[rb + 64];
  const float k1 = qkv[rb + HID_],     k2 = qkv[rb + HID_ + 64];
  const float v1 = qkv[rb + 2 * HID_], v2 = qkv[rb + 2 * HID_ + 64];

  const size_t headBase = (size_t)(b * NH_ + h) * S_ * HD_;
  const int d2 = d + 64;

  // Q/K operand tiles: lane from s, elem from d.
  const size_t qk0 = headBase + ((size_t)((s >> 4) * 4 + (d  >> 5)) << 9) + op_lane(s, d ) * 16 + op_elem(d );
  const size_t qk1 = headBase + ((size_t)((s >> 4) * 4 + (d2 >> 5)) << 9) + op_lane(s, d2) * 16 + op_elem(d2);
  Qop[qk0] = (__bf16)(q1 * cs - q2 * sn);
  Qop[qk1] = (__bf16)(q2 * cs + q1 * sn);
  Kop[qk0] = (__bf16)(k1 * cs - k2 * sn);
  Kop[qk1] = (__bf16)(k2 * cs + k1 * sn);

  // V B-operand tiles: lane from d, elem from s.
  const size_t v0 = headBase + ((size_t)((s >> 5) * 8 + (d  >> 4)) << 9) + op_lane(d,  s) * 16 + op_elem(s);
  const size_t vD = headBase + ((size_t)((s >> 5) * 8 + (d2 >> 4)) << 9) + op_lane(d2, s) * 16 + op_elem(s);
  Vop[v0] = (__bf16)v1;
  Vop[vD] = (__bf16)v2;
}

// ---------------------------------------------------------------------------
// Causal flash attention. Block = 128 threads (4 waves) = one (b,h,64-row qblk).
// Wave q-tile = 16 rows. K consumed in 32-col blocks: 8 WMMA (QK^T) + 8 (PV).
// Double-buffered async K/V staging; output written in A-operand layout.
// ---------------------------------------------------------------------------
__global__ __launch_bounds__(128, 1)
void flash_attn_kernel(const __bf16* __restrict__ Qop,
                       const __bf16* __restrict__ Kop,
                       const __bf16* __restrict__ Vop,
                       __bf16* __restrict__ attnA) {
  __shared__ __align__(32) __bf16 ldsK[2][8 * 512];
  __shared__ __align__(32) __bf16 ldsV[2][8 * 512];
  __shared__ __align__(32) __bf16 ldsP[4 * 512];     // per-wave P, A-op layout

  const int t    = threadIdx.x;
  const int wv   = t >> 5;
  const int lane = t & 31;

  const int qblk = blockIdx.x % (S_ / 64);
  const int h    = (blockIdx.x / (S_ / 64)) % NH_;
  const int b    = blockIdx.x / ((S_ / 64) * NH_);
  const int q0   = qblk * 64;
  const size_t headBase = (size_t)(b * NH_ + h) * S_ * HD_;

  // Per-wave Q operand (16 rows x 128 d), direct coalesced global b128 loads.
  bf16x16 aq[4];
#pragma unroll
  for (int c = 0; c < 4; ++c)
    aq[c] = *(const bf16x16*)(Qop + headBase
             + ((size_t)(((q0 >> 4) + wv) * 4 + c) << 9) + lane * 16);

  float m_i[8], l_i[8];
  f32x8 acc[8];
#pragma unroll
  for (int r = 0; r < 8; ++r) { m_i[r] = NEG_; l_i[r] = 0.0f; }
#pragma unroll
  for (int j = 0; j < 8; ++j) acc[j] = (f32x8){};

  const int nkb      = (q0 + 64) / 32;
  const int myRowMax = q0 + wv * 16 + 15;

  // ---- initial stage of kb=0 ----
#if HAVE_ASYNC_LDS
#pragma unroll
  for (int i = 0; i < 4; ++i) {
    const int c = t + 128 * i;
    async_cp16((const __bf16*)(((const float4*)(Kop + headBase)) + c),
               (__bf16*)(((float4*)ldsK[0]) + c));
    async_cp16((const __bf16*)(((const float4*)(Vop + headBase)) + c),
               (__bf16*)(((float4*)ldsV[0]) + c));
  }
  wait_async0();
#else
#pragma unroll
  for (int i = 0; i < 4; ++i) {
    const int c = t + 128 * i;
    ((float4*)ldsK[0])[c] = ((const float4*)(Kop + headBase))[c];
    ((float4*)ldsV[0])[c] = ((const float4*)(Vop + headBase))[c];
  }
#endif
  __syncthreads();

  for (int kb = 0; kb < nkb; ++kb) {
    const int cur = kb & 1, nxt = cur ^ 1;
    const bool more = (kb + 1 < nkb);
    const __bf16* ksrcN = Kop + headBase + ((size_t)((kb + 1) * 8) << 9);
    const __bf16* vsrcN = Vop + headBase + ((size_t)((kb + 1) * 8) << 9);

#if HAVE_ASYNC_LDS
    if (more) {
#pragma unroll
      for (int i = 0; i < 4; ++i) {
        const int c = t + 128 * i;
        async_cp16((const __bf16*)(((const float4*)ksrcN) + c),
                   (__bf16*)(((float4*)ldsK[nxt]) + c));
        async_cp16((const __bf16*)(((const float4*)vsrcN) + c),
                   (__bf16*)(((float4*)ldsV[nxt]) + c));
      }
    }
#else
    float4 rk[4], rv[4];
    if (more) {
#pragma unroll
      for (int i = 0; i < 4; ++i) {
        const int c = t + 128 * i;
        rk[i] = ((const float4*)ksrcN)[c];
        rv[i] = ((const float4*)vsrcN)[c];
      }
    }
#endif
    if (kb + 2 < nkb) {
      __builtin_prefetch(Kop + headBase + ((size_t)((kb + 2) * 8) << 9) + t * 32, 0, 1);
      __builtin_prefetch(Vop + headBase + ((size_t)((kb + 2) * 8) << 9) + t * 32, 0, 1);
    }

    if (kb * 32 <= myRowMax) {                         // uniform per wave
      // ---- S = Q K^T, two 16x16 tiles over 32 k-cols ----
      f32x8 s0 = (f32x8){}, s1 = (f32x8){};
#pragma unroll
      for (int c = 0; c < 4; ++c) {
        const bf16x16 b0 = *(const bf16x16*)&ldsK[cur][(c << 9) + lane * 16];
        s0 = wmma_bf16(aq[c], b0, s0);
        const bf16x16 b1 = *(const bf16x16*)&ldsK[cur][((4 + c) << 9) + lane * 16];
        s1 = wmma_bf16(aq[c], b1, s1);
      }

      // ---- scale + causal mask + online softmax ----
      const int colBase = kb * 32 + (lane & 15);
      const int rowG    = q0 + wv * 16 + ((lane >> 4) << 3);
      float mnew[8], alpha[8];
#pragma unroll
      for (int r = 0; r < 8; ++r) {
        float a0 = s0[r] * SCALE_;
        float a1 = s1[r] * SCALE_;
        const int rg = rowG + r;
        if (colBase > rg)      a0 = NEG_;
        if (colBase + 16 > rg) a1 = NEG_;
        s0[r] = a0; s1[r] = a1;
        float v = fmaxf(a0, a1);
        v = fmaxf(v, __shfl_xor(v, 1, 32));
        v = fmaxf(v, __shfl_xor(v, 2, 32));
        v = fmaxf(v, __shfl_xor(v, 4, 32));
        v = fmaxf(v, __shfl_xor(v, 8, 32));
        mnew[r] = fmaxf(m_i[r], v);
      }
#pragma unroll
      for (int r = 0; r < 8; ++r) {
        alpha[r] = __expf(m_i[r] - mnew[r]);
        m_i[r]   = mnew[r];
        const float p0 = __expf(s0[r] - mnew[r]);
        const float p1 = __expf(s1[r] - mnew[r]);
        s0[r] = p0; s1[r] = p1;
        float rs = p0 + p1;
        rs += __shfl_xor(rs, 1, 32);
        rs += __shfl_xor(rs, 2, 32);
        rs += __shfl_xor(rs, 4, 32);
        rs += __shfl_xor(rs, 8, 32);
        l_i[r] = l_i[r] * alpha[r] + rs;
      }
#pragma unroll
      for (int j = 0; j < 8; ++j)
#pragma unroll
        for (int r = 0; r < 8; ++r) acc[j][r] *= alpha[r];

      // ---- store P straight into A-operand layout (per-wave LDS tile) ----
      __bf16* pw = &ldsP[wv * 512];
      const int hiP = (lane >> 3) & 1;       // from n = lane&15 (same for n+16)
#pragma unroll
      for (int r = 0; r < 8; ++r) {
        const int m = r + ((lane >> 4) << 3);
        const int lane_p = m + (hiP << 4);
        pw[lane_p * 16 + (lane & 7)]     = (__bf16)s0[r];
        pw[lane_p * 16 + (lane & 7) + 8] = (__bf16)s1[r];
      }
      // Wave-local LDS RAW (no block barrier: other waves may be skipping).
      asm volatile("s_wait_dscnt 0x0" ::: "memory");
      const bf16x16 ap = *(const bf16x16*)&pw[lane * 16];

      // ---- O += P V over 8 d-tiles ----
#pragma unroll
      for (int jt = 0; jt < 8; ++jt) {
        const bf16x16 bv = *(const bf16x16*)&ldsV[cur][(jt << 9) + lane * 16];
        acc[jt] = wmma_bf16(ap, bv, acc[jt]);
      }
    }

#if HAVE_ASYNC_LDS
    if (more) wait_async0();
#else
    if (more) {
#pragma unroll
      for (int i = 0; i < 4; ++i) {
        const int c = t + 128 * i;
        ((float4*)ldsK[nxt])[c] = rk[i];
        ((float4*)ldsV[nxt])[c] = rv[i];
      }
    }
#endif
    __syncthreads();
  }

  // ---- normalize; write output as A-operand tiles for the out-projection ----
  float invl[8];
#pragma unroll
  for (int r = 0; r < 8; ++r) invl[r] = 1.0f / l_i[r];
  const int KBo = HID_ >> 5;                 // 64 k-blocks in out-proj
  const int mTokBase = b * S_ + q0 + wv * 16;
#pragma unroll
  for (int jt = 0; jt < 8; ++jt)
#pragma unroll
    for (int r = 0; r < 8; ++r) {
      const int m  = mTokBase + r + ((lane >> 4) << 3);
      const int cH = h * HD_ + jt * 16 + (lane & 15);
      const size_t dst = ((size_t)((m >> 4) * KBo + (cH >> 5)) << 9)
                       + op_lane(m, cH) * 16 + op_elem(cH);
      attnA[dst] = (__bf16)(acc[jt][r] * invl[r]);
    }
}

// ---------------------------------------------------------------------------
// Host-side launch
// ---------------------------------------------------------------------------
extern "C" void kernel_launch(void* const* d_in, const int* in_sizes, int n_in,
                              void* d_out, int out_size, void* d_ws, size_t ws_size,
                              hipStream_t stream) {
  const float* hs   = (const float*)d_in[0];   // (B,S,HID) f32
  const int*   pos  = (const int*)d_in[1];     // (S,) i32
  const float* wqkv = (const float*)d_in[2];   // (HID, 3*HID) f32
  const float* wo   = (const float*)d_in[3];   // (HID, HID) f32
  float*       out  = (float*)d_out;           // (B,S,HID) f32

  const int MT = B_ * S_;                      // 4096 tokens

  char* p = (char*)d_ws;
  __bf16* hsA   = (__bf16*)p; p += (size_t)MT * HID_ * 2;            // A-op tiles
  __bf16* wqkvB = (__bf16*)p; p += (size_t)HID_ * 3 * HID_ * 2;      // B-op tiles
  __bf16* woB   = (__bf16*)p; p += (size_t)HID_ * HID_ * 2;          // B-op tiles
  float*  qkvf  = (float*)p;  p += (size_t)MT * 3 * HID_ * 4;        // f32 row-major
  __bf16* Qop   = (__bf16*)p; p += (size_t)B_ * NH_ * S_ * HD_ * 2;
  __bf16* Kop   = (__bf16*)p; p += (size_t)B_ * NH_ * S_ * HD_ * 2;
  __bf16* Vop   = (__bf16*)p; p += (size_t)B_ * NH_ * S_ * HD_ * 2;
  __bf16* attnA = (__bf16*)p; p += (size_t)MT * HID_ * 2;            // A-op tiles

  // 1) swizzling casts to bf16 operand layouts
  swizzle_opA_kernel<<<4096, 256, 0, stream>>>(hs,   hsA,   MT,   HID_);
  swizzle_opB_kernel<<<4096, 256, 0, stream>>>(wqkv, wqkvB, HID_, 3 * HID_);
  swizzle_opB_kernel<<<4096, 256, 0, stream>>>(wo,   woB,   HID_, HID_);

  // 2) QKV GEMM + clip: (4096 x 2048) x (2048 x 6144) -> f32
  gemm_bf16_wmma<<<dim3(3 * HID_ / 128, MT / 128), 128, 0, stream>>>(
      hsA, wqkvB, qkvf, MT, 3 * HID_, HID_, /*doClip=*/1);

  // 3) RoPE + head split into operand-layout Q/K/V
  {
    const int nthreads = B_ * S_ * NH_ * 64;
    rope_split_kernel<<<(nthreads + 255) / 256, 256, 0, stream>>>(
        qkvf, pos, Qop, Kop, Vop);
  }

  // 4) causal flash attention -> A-operand-layout activations
  flash_attn_kernel<<<B_ * NH_ * (S_ / 64), 128, 0, stream>>>(Qop, Kop, Vop, attnA);

  // 5) output projection: (4096 x 2048) x (2048 x 2048) -> d_out f32
  gemm_bf16_wmma<<<dim3(HID_ / 128, MT / 128), 128, 0, stream>>>(
      attnA, woB, out, MT, HID_, HID_, /*doClip=*/0);
}
